// ScaledDotProductAttention_KNN_28819230556612
// MI455X (gfx1250) — matrix operand order repaired and checked by reading, not compiled, not run on hardware
//
#include <hip/hip_runtime.h>
#include <stdint.h>

#define NKEYS 32
#define DIM   256
#define WAVES_PER_BLOCK 4
#define THREADS (WAVES_PER_BLOCK * 32)

// Per-wave LDS layout (floats): K tile | q | attn
#define KF    (NKEYS * DIM)             // 8192 floats = 32 KB
#define QOFF  KF
#define AOFF  (KF + DIM)
#define WAVE_LDS_F (KF + DIM + NKEYS)   // 8480 floats = 33920 B (16B aligned)

typedef __attribute__((ext_vector_type(16))) __bf16   v16bf;
typedef __attribute__((ext_vector_type(8)))  float    v8f;
typedef __attribute__((ext_vector_type(4)))  float    v4f;
typedef __attribute__((ext_vector_type(4)))  uint32_t v4u;

union V16BF {
  v16bf v;
  uint32_t u[8];
};

// Two f32 -> packed bf16x2 (truncating) in ONE v_perm_b32:
// result = { hi[31:16], lo[31:16] }
static __device__ inline uint32_t pk2bf(uint32_t lo, uint32_t hi) {
  return __builtin_amdgcn_perm(hi, lo, 0x07060302u);
}

// CDNA5 async global->LDS copy, 16B per lane (512B per wave per issue).
// GVS addressing: mem = SADDR(64) + VADDR(32); VDST VGPR holds LDS byte addr.
static __device__ inline void async_copy_b128(uint32_t lds_addr,
                                              const void* gbase,
                                              uint32_t goff) {
  asm volatile("global_load_async_to_lds_b128 %0, %1, %2"
               :
               : "v"(lds_addr), "v"(goff), "s"(gbase)
               : "memory");
}

static __device__ inline void wait_async0() {
#if __has_builtin(__builtin_amdgcn_s_wait_asynccnt)
  __builtin_amdgcn_s_wait_asynccnt(0);
#else
  asm volatile("s_wait_asynccnt 0" ::: "memory");
#endif
}

extern "C" __global__ void __launch_bounds__(THREADS)
sdpa_knn_kernel(const float* __restrict__ qg,
                const float* __restrict__ kg,
                float* __restrict__ outg,
                int nBatch) {
  extern __shared__ float lds[];
  const int lane = threadIdx.x & 31;
  const int wave = threadIdx.x >> 5;
  int batch = blockIdx.x * WAVES_PER_BLOCK + wave;
  if (batch >= nBatch) batch = nBatch - 1;  // keep EXEC all-ones for WMMA

  float* kl = lds + wave * WAVE_LDS_F;  // K[32][256] f32
  float* ql = kl + QOFF;                // q[256] f32
  float* al = kl + AOFF;                // scores/attn[32]

  const float* kb = kg + (size_t)batch * (NKEYS * DIM);
  const float* qb = qg + (size_t)batch * DIM;

  // ---- stage K (32 KB) + q (1 KB) into LDS with async b128 copies ----
  const uint32_t klAddr = (uint32_t)(uintptr_t)kl;  // flat low-32 = LDS offset
  const uint32_t qlAddr = (uint32_t)(uintptr_t)ql;
#pragma unroll
  for (int i = 0; i < 64; ++i) {
    const uint32_t off = (uint32_t)(i * 512 + lane * 16);
    async_copy_b128(klAddr + off, kb, off);
  }
  {
    const uint32_t off = (uint32_t)(lane * 16);
    async_copy_b128(qlAddr + off, qb, off);
    async_copy_b128(qlAddr + off + 512, qb, off + 512);
  }
  wait_async0();
  __builtin_amdgcn_wave_barrier();

  // ---- phase 1: scores[32] = K @ q via v_wmma_f32_16x16x32_bf16 ----
  // A (16x32 bf16): lane m = lane&15, half h = lane>>4 holds
  //   vgpr0-3: k = 8h + 0..7 ; vgpr4-7: k = 16 + 8h + 0..7
  // B (32x16 bf16): lane col n = lane&15; same k layout as A
  // B columns replicated with q -> every D column holds the score vector.
  const int m = lane & 15;
  const int h = lane >> 4;

  v8f acc0 = {}, acc1 = {};
#pragma unroll
  for (int c = 0; c < 8; ++c) {
    // B fragment (shared by both key tiles): q[c*32 + 8h + 0..7], q[c*32 + 16 + 8h + 0..7]
    V16BF B;
    {
      const v4u* qp = (const v4u*)(ql + c * 32 + h * 8);
      const v4u q0 = qp[0], q1 = qp[1], q2 = qp[4], q3 = qp[5];
      B.u[0] = pk2bf(q0.x, q0.y);
      B.u[1] = pk2bf(q0.z, q0.w);
      B.u[2] = pk2bf(q1.x, q1.y);
      B.u[3] = pk2bf(q1.z, q1.w);
      B.u[4] = pk2bf(q2.x, q2.y);
      B.u[5] = pk2bf(q2.z, q2.w);
      B.u[6] = pk2bf(q3.x, q3.y);
      B.u[7] = pk2bf(q3.z, q3.w);
    }
#pragma unroll
    for (int t = 0; t < 2; ++t) {
      const float* arow = kl + (t * 16 + m) * DIM + c * 32 + h * 8;
      const v4u* ap0 = (const v4u*)(arow);       // k = 8h + 0..7
      const v4u* ap1 = (const v4u*)(arow + 16);  // k = 16 + 8h + 0..7
      const v4u a0 = ap0[0], a1 = ap0[1];
      const v4u a2 = ap1[0], a3 = ap1[1];
      V16BF A;
      A.u[0] = pk2bf(a0.x, a0.y);
      A.u[1] = pk2bf(a0.z, a0.w);
      A.u[2] = pk2bf(a1.x, a1.y);
      A.u[3] = pk2bf(a1.z, a1.w);
      A.u[4] = pk2bf(a2.x, a2.y);
      A.u[5] = pk2bf(a2.z, a2.w);
      A.u[6] = pk2bf(a3.x, a3.y);
      A.u[7] = pk2bf(a3.z, a3.w);
      if (t == 0)
        acc0 = __builtin_amdgcn_wmma_f32_16x16x32_bf16(
            false, A.v, false, B.v, (short)0, acc0, false, false);
      else
        acc1 = __builtin_amdgcn_wmma_f32_16x16x32_bf16(
            false, A.v, false, B.v, (short)0, acc1, false, false);
    }
  }
  // D layout: vgpr v -> M=v (lanes 0-15) / M=v+8 (lanes 16-31); cols equal.
  if (lane == 0) {
#pragma unroll
    for (int v = 0; v < 8; ++v) {
      al[v] = acc0[v];
      al[16 + v] = acc1[v];
    }
  }
  if (lane == 16) {
#pragma unroll
    for (int v = 0; v < 8; ++v) {
      al[8 + v] = acc0[v];
      al[24 + v] = acc1[v];
    }
  }
  __builtin_amdgcn_wave_barrier();

  // ---- softmax over 32 keys (one score per lane, wave32 reductions) ----
  const float s = al[lane] * 0.0625f;  // 1/sqrt(256)
  float mx = s;
#pragma unroll
  for (int d = 16; d > 0; d >>= 1) mx = fmaxf(mx, __shfl_xor(mx, d, 32));
  const float e = __expf(s - mx);
  float sum = e;
#pragma unroll
  for (int d = 16; d > 0; d >>= 1) sum += __shfl_xor(sum, d, 32);
  al[lane] = e / sum;
  __builtin_amdgcn_wave_barrier();

  // ---- phase 2: context = attn @ K in exact f32; out = q + context ----
  v4f c0 = {}, c1 = {};
#pragma unroll 8
  for (int kk = 0; kk < NKEYS; ++kk) {
    const float w = al[kk];  // uniform broadcast from LDS
    const v4f* row = (const v4f*)(kl + kk * DIM + lane * 8);
    c0 += w * row[0];
    c1 += w * row[1];
  }
  const v4f* qv = (const v4f*)(ql + lane * 8);
  const v4f o0 = qv[0] + c0;
  const v4f o1 = qv[1] + c1;
  float* ob = outg + (size_t)batch * DIM + lane * 8;
  *((v4f*)ob)     = o0;
  *((v4f*)ob + 1) = o1;
}

extern "C" void kernel_launch(void* const* d_in, const int* in_sizes, int n_in,
                              void* d_out, int out_size, void* d_ws, size_t ws_size,
                              hipStream_t stream) {
  const float* q = (const float*)d_in[0];  // [N,1,256] f32
  const float* k = (const float*)d_in[1];  // [N,32,256] f32
  float* out = (float*)d_out;              // [N,1,256] f32
  const int nBatch = in_sizes[0] / DIM;    // 32768
  const int blocks = (nBatch + WAVES_PER_BLOCK - 1) / WAVES_PER_BLOCK;
  const size_t shmem = (size_t)WAVES_PER_BLOCK * WAVE_LDS_F * sizeof(float);
  sdpa_knn_kernel<<<blocks, THREADS, shmem, stream>>>(q, k, out, nBatch);
}
